// _Expert_11871289606688
// MI455X (gfx1250) — matrix operand order repaired and checked by reading, not compiled, not run on hardware
//
#include <hip/hip_runtime.h>
#include <hip/hip_bf16.h>

// ---------------------------------------------------------------------------
// MoE expert GEMM for MI455X (gfx1250), wave32 + WMMA.
// out[t, :] = inp[t, :] @ weight[e].T + bias[e],  e from fwd_expert_count.
// fp32 in/out; math via bf16x3 split (Ah*Bh + Ah*Bl + Al*Bh) on
// v_wmma_f32_16x16x32_bf16 with f32 accumulation (~fp32 accuracy).
//
// Tile: BM=256 x BN=128 x BK=32, 256 threads (8 waves, 4(M) x 2(N)),
// 64x64 per wave -> 48 wmma per 32 ds_load_b128 per K-step.
// LDS: double-buffered (one barrier per K-step), layout [K/8][row][8 half]
// (fragment-granular, 16B aligned, conflict-free half-wave access),
// 96 KB total (2 x (A hi/lo 32K + B hi/lo 16K)); WGP allows 320 KB.
// ---------------------------------------------------------------------------

#define N_EXPERTS 8
#define DIM_IN    2048
#define DIM_OUT   8192
#define N_TOK     16384

#define BM 256
#define BN 128
#define BK 32
#define KSTEPS (DIM_IN / BK)

#define A_STAGE (4 * BM * 8)   // halfwords per A plane per buffer
#define B_STAGE (4 * BN * 8)   // halfwords per B plane per buffer

typedef __attribute__((ext_vector_type(16))) __bf16 bf16x16;
typedef __attribute__((ext_vector_type(8)))  float  f32x8;

union Frag { uint4 u[2]; bf16x16 v; };

// Truncating hi/lo split of two fp32 into packed bf16 dwords.
// hi = top 16 bits of a; lo = bf16_trunc(a - hi). lo corrects hi's
// truncation, so combined precision ~16 mantissa bits.
__device__ __forceinline__ void split2(float a0, float a1,
                                       unsigned& hp, unsigned& lp) {
  unsigned u0 = __float_as_uint(a0), u1 = __float_as_uint(a1);
  unsigned h0 = u0 & 0xFFFF0000u, h1 = u1 & 0xFFFF0000u;
  float l0 = a0 - __uint_as_float(h0);
  float l1 = a1 - __uint_as_float(h1);
  hp = (h0 >> 16) | h1;
  lp = (__float_as_uint(l0) >> 16) | (__float_as_uint(l1) & 0xFFFF0000u);
}

__global__ __launch_bounds__(256) void moe_expert_gemm(
    const float* __restrict__ inp, const int* __restrict__ cnt,
    const float* __restrict__ weight, const float* __restrict__ bias,
    float* __restrict__ out) {
  // Ping-pong buffered planes: [buf][g][row][8 halfwords].
  __shared__ __align__(16) unsigned short sAh[2 * A_STAGE];
  __shared__ __align__(16) unsigned short sAl[2 * A_STAGE];
  __shared__ __align__(16) unsigned short sBh[2 * B_STAGE];
  __shared__ __align__(16) unsigned short sBl[2 * B_STAGE];

  const int rowBase = blockIdx.y * BM;
  const int colBase = blockIdx.x * BN;

  // Resolve expert segment for this row tile (E=8: cheap uniform scan).
  int segStart = 0, segEnd = 0, e = 0, off = 0;
#pragma unroll
  for (int i = 0; i < N_EXPERTS; ++i) {
    int c = cnt[i];
    if (rowBase >= off && rowBase < off + c) { e = i; segStart = off; segEnd = off + c; }
    off += c;
  }
  if (segEnd <= segStart) return;  // uniform over block

  const int tid = threadIdx.x;

  // A loader: 1 thread per row, 32 K-floats (8 float4) per step.
  const float* aPtr = inp + (size_t)(rowBase + tid) * DIM_IN;
  const bool aValid = (rowBase + tid) < segEnd;
  // B loader: 2 threads per row, 16 K-floats (4 float4) per step.
  const int bRow = tid >> 1;
  const int bKh  = (tid & 1) * 16;
  const float* bPtr = weight + (size_t)e * DIM_OUT * DIM_IN +
                      (size_t)(colBase + bRow) * DIM_IN + bKh;

  float4 aStage[8], bStage[4];
#pragma unroll
  for (int q = 0; q < 8; ++q)
    aStage[q] = aValid ? *(const float4*)(aPtr + q * 4) : make_float4(0.f, 0.f, 0.f, 0.f);
#pragma unroll
  for (int q = 0; q < 4; ++q)
    bStage[q] = *(const float4*)(bPtr + q * 4);

  const int lane = tid & 31;
  const int wave = tid >> 5;
  const int wm   = wave >> 1;              // 0..3 -> 64-row strip
  const int wn   = wave & 1;               // 0..1 -> 64-col strip
  const int l15  = lane & 15;
  const int g0A  = (lane < 16) ? 0 : 2;    // A frag: K groups g0A, g0A+1
  const int g0B  = (lane < 16) ? 0 : 2;    // B frag: K groups g0B, g0B+1

  f32x8 acc[4][4];
#pragma unroll
  for (int i = 0; i < 4; ++i)
#pragma unroll
    for (int j = 0; j < 4; ++j)
#pragma unroll
      for (int r = 0; r < 8; ++r) acc[i][j][r] = 0.f;

  for (int ks = 0; ks < KSTEPS; ++ks) {
    const int bo  = (ks & 1) ? A_STAGE : 0;   // A-plane buffer offset
    const int bob = (ks & 1) ? B_STAGE : 0;   // B-plane buffer offset

    // ---- split staged tile into bf16 hi/lo planes, store to LDS ----
    {
      unsigned hp[16], lp[16];
#pragma unroll
      for (int q = 0; q < 8; ++q) {
        split2(aStage[q].x, aStage[q].y, hp[q * 2], lp[q * 2]);
        split2(aStage[q].z, aStage[q].w, hp[q * 2 + 1], lp[q * 2 + 1]);
      }
#pragma unroll
      for (int g = 0; g < 4; ++g) {
        const int o = bo + (g * BM + tid) * 8;
        *(uint4*)&sAh[o] = make_uint4(hp[g * 4], hp[g * 4 + 1], hp[g * 4 + 2], hp[g * 4 + 3]);
        *(uint4*)&sAl[o] = make_uint4(lp[g * 4], lp[g * 4 + 1], lp[g * 4 + 2], lp[g * 4 + 3]);
      }
    }
    {
      unsigned hp[8], lp[8];
#pragma unroll
      for (int q = 0; q < 4; ++q) {
        split2(bStage[q].x, bStage[q].y, hp[q * 2], lp[q * 2]);
        split2(bStage[q].z, bStage[q].w, hp[q * 2 + 1], lp[q * 2 + 1]);
      }
#pragma unroll
      for (int g = 0; g < 2; ++g) {
        const int o = bob + ((bKh / 8 + g) * BN + bRow) * 8;
        *(uint4*)&sBh[o] = make_uint4(hp[g * 4], hp[g * 4 + 1], hp[g * 4 + 2], hp[g * 4 + 3]);
        *(uint4*)&sBl[o] = make_uint4(lp[g * 4], lp[g * 4 + 1], lp[g * 4 + 2], lp[g * 4 + 3]);
      }
    }
    __syncthreads();  // single barrier per K-step (ping-pong buffers)

    // ---- prefetch next K-step (overlaps with WMMA below) ----
    if (ks + 1 < KSTEPS) {
      const float* ap = aPtr + (size_t)(ks + 1) * BK;
      const float* bp = bPtr + (size_t)(ks + 1) * BK;
#pragma unroll
      for (int q = 0; q < 8; ++q)
        aStage[q] = aValid ? *(const float4*)(ap + q * 4) : make_float4(0.f, 0.f, 0.f, 0.f);
#pragma unroll
      for (int q = 0; q < 4; ++q)
        bStage[q] = *(const float4*)(bp + q * 4);
    }

    // ---- A fragments (16x32 bf16, lane half h holds k 16h..16h+15, same order as B) ----
    Frag ah[4], al[4];
#pragma unroll
    for (int tm = 0; tm < 4; ++tm) {
      const int row = wm * 64 + tm * 16 + l15;
      ah[tm].u[0] = *(const uint4*)&sAh[bo + ((g0A)*BM + row) * 8];
      ah[tm].u[1] = *(const uint4*)&sAh[bo + ((g0A + 1) * BM + row) * 8];
      al[tm].u[0] = *(const uint4*)&sAl[bo + ((g0A)*BM + row) * 8];
      al[tm].u[1] = *(const uint4*)&sAl[bo + ((g0A + 1) * BM + row) * 8];
    }

    // ---- B fragments + 48 WMMA ----
#pragma unroll
    for (int tn = 0; tn < 4; ++tn) {
      const int n = wn * 64 + tn * 16 + l15;
      Frag bh, bl;
      bh.u[0] = *(const uint4*)&sBh[bob + ((g0B)*BN + n) * 8];
      bh.u[1] = *(const uint4*)&sBh[bob + ((g0B + 1) * BN + n) * 8];
      bl.u[0] = *(const uint4*)&sBl[bob + ((g0B)*BN + n) * 8];
      bl.u[1] = *(const uint4*)&sBl[bob + ((g0B + 1) * BN + n) * 8];
#pragma unroll
      for (int tm = 0; tm < 4; ++tm) {
        acc[tm][tn] = __builtin_amdgcn_wmma_f32_16x16x32_bf16(
            false, ah[tm].v, false, bh.v, (short)0, acc[tm][tn], false, false);
        acc[tm][tn] = __builtin_amdgcn_wmma_f32_16x16x32_bf16(
            false, al[tm].v, false, bh.v, (short)0, acc[tm][tn], false, false);
        acc[tm][tn] = __builtin_amdgcn_wmma_f32_16x16x32_bf16(
            false, ah[tm].v, false, bl.v, (short)0, acc[tm][tn], false, false);
      }
    }
    // no trailing barrier: next iteration writes the other buffer; reuse of
    // this buffer is fenced by the next iteration's barrier + program order.
  }

  // Epilogue: C/D layout — VGPR r: lanes 0-15 => M=r, lanes 16-31 => M=r+8.
  const bool fullTile = (rowBase + BM) <= segEnd;  // block-uniform
  if (fullTile) {
#pragma unroll
    for (int tn = 0; tn < 4; ++tn) {
      const int col = colBase + wn * 64 + tn * 16 + l15;
      const float bv = bias[e * DIM_OUT + col];
#pragma unroll
      for (int tm = 0; tm < 4; ++tm) {
        const int rbase = rowBase + wm * 64 + tm * 16 + ((lane < 16) ? 0 : 8);
#pragma unroll
        for (int r = 0; r < 8; ++r)
          out[(size_t)(rbase + r) * DIM_OUT + col] = acc[tm][tn][r] + bv;
      }
    }
  } else {
#pragma unroll
    for (int tn = 0; tn < 4; ++tn) {
      const int col = colBase + wn * 64 + tn * 16 + l15;
      const float bv = bias[e * DIM_OUT + col];
#pragma unroll
      for (int tm = 0; tm < 4; ++tm) {
        const int rbase = rowBase + wm * 64 + tm * 16 + ((lane < 16) ? 0 : 8);
#pragma unroll
        for (int r = 0; r < 8; ++r) {
          const int row = rbase + r;
          if (row < segEnd)
            out[(size_t)row * DIM_OUT + col] = acc[tm][tn][r] + bv;
        }
      }
    }
  }
}

extern "C" void kernel_launch(void* const* d_in, const int* in_sizes, int n_in,
                              void* d_out, int out_size, void* d_ws, size_t ws_size,
                              hipStream_t stream) {
  const float* inp    = (const float*)d_in[0];
  const int*   cnt    = (const int*)d_in[1];
  const float* weight = (const float*)d_in[2];
  const float* bias   = (const float*)d_in[3];
  float*       out    = (float*)d_out;

  dim3 grid(DIM_OUT / BN, N_TOK / BM);  // (64, 64)
  moe_expert_gemm<<<grid, 256, 0, stream>>>(inp, cnt, weight, bias, out);
}
